// NeRFRenderer_81329500717186
// MI455X (gfx1250) — compile-verified, matrix operand order checked by reading, not run on hardware
//
#include <hip/hip_runtime.h>
#include <hip/hip_bf16.h>
#include <math.h>

// NeRF density-MLP volume renderer for gfx1250 (MI455X), wave32 + WMMA.
//
// Transposed formulation: per 16-sample tile, compute
//   h^T [hidden x 16] = W^T (A, stationary in VGPRs) x act^T (B)
// Layers 1-2 use f16 C/D accumulators: the 16-bit C/D layout is half-for-half
// identical to the next layer's f16 B-operand layout, so the layer handoff is
// exactly 8x v_pk_max_num_f16 (relu, forced via inline asm to skip clang's
// canonicalize) + a zero-cost register concat (shufflevector).
// Layer 3 accumulates in f32 (feeds trunc_exp).

typedef __attribute__((ext_vector_type(16))) _Float16     v16h;
typedef __attribute__((ext_vector_type(8)))  _Float16     v8h;
typedef __attribute__((ext_vector_type(8)))  float        v8f;
typedef __attribute__((ext_vector_type(8)))  unsigned int v8u;

#define BOUNDV   1.0f
#define MIN_NEAR 0.05f
#define N_RAYS   16384
#define HIDDEN   64
#define WAVES_PER_BLOCK 8
#define NUM_BLOCKS      512
#define TOTAL_WAVES     (WAVES_PER_BLOCK * NUM_BLOCKS)

static __device__ __forceinline__ v8h wmma_h(v16h a, v16h b, v8h c) {
  // f16 C/D variant: (neg_a, A, neg_b, B, c_mod, C, reuse_a, reuse_b)
  return __builtin_amdgcn_wmma_f16_16x16x32_f16(false, a, false, b, (short)0, c,
                                                false, false);
}
static __device__ __forceinline__ v8f wmma_f(v16h a, v16h b, v8f c) {
  return __builtin_amdgcn_wmma_f32_16x16x32_f16(false, a, false, b, (short)0, c,
                                                false, false);
}
static __device__ __forceinline__ v16h cat16(v8h lo, v8h hi) {
  return __builtin_shufflevector(lo, hi, 0, 1, 2, 3, 4, 5, 6, 7,
                                 8, 9, 10, 11, 12, 13, 14, 15);
}
// relu on a concatenated B fragment: exactly 8 packed f16 max ops.
static __device__ __forceinline__ v16h relu_cat(v8h lo, v8h hi) {
  union { v16h h; v8u u; } c;
  c.h = cat16(lo, hi);
  #pragma unroll
  for (int i = 0; i < 8; i++) {
    unsigned int t = c.u[i];
    asm("v_pk_max_num_f16 %0, %1, 0" : "=v"(t) : "v"(t));
    c.u[i] = t;
  }
  return c.h;
}

__global__ __launch_bounds__(256)
void nerf_render_kernel(const float* __restrict__ rays_o,
                        const float* __restrict__ rays_d,
                        const float* __restrict__ W1, const float* __restrict__ b1,
                        const float* __restrict__ W2, const float* __restrict__ b2,
                        const float* __restrict__ W3, const float* __restrict__ b3,
                        const int* __restrict__ nsamp,
                        float* __restrict__ out)
{
  const int lane  = threadIdx.x & 31;
  const int waveG = blockIdx.x * (blockDim.x >> 5) + (threadIdx.x >> 5);
  const int m16   = lane & 15;   // row within 16-tile (hidden idx) / sample idx
  const int chunk = lane >> 4;   // selects K sub-chunk in A/B f16 layouts

  const int   S      = nsamp[0];
  const int   ntiles = S >> 4;
  const float step   = 2.0f * BOUNDV * 1.7320508075688772f / (float)S;
  const float b3s    = b3[0];

  // ---------------- Stationary A-fragments (weights), loaded once per wave.
  // f16 A layout: lane holds row M = 16*tile + m16;
  // half h -> K = 8*chunk + (h<8 ? h : h+8).
  v16h A1[4];      // W1aug^T: K=0..2 -> W1 rows, K=3 -> b1, K>=4 -> 0
  v16h A2[4][2];   // W2^T: 4 M-tiles x 2 K-steps
  v16h A3[2];      // W3^T lives in row M==0 only
  v8h  C2b[4];     // L2 initial accumulator = b2 in f16 C layout:
                   //   half h -> b2[16i + 8*chunk + h]

  #pragma unroll
  for (int i = 0; i < 4; i++) { A1[i] = {}; }
  if (chunk == 0) {            // chunk 1 lanes cover K in {8..15,24..31} -> all zero
    #pragma unroll
    for (int i = 0; i < 4; i++) {
      int M = 16 * i + m16;
      A1[i][0] = (_Float16)W1[0 * HIDDEN + M];
      A1[i][1] = (_Float16)W1[1 * HIDDEN + M];
      A1[i][2] = (_Float16)W1[2 * HIDDEN + M];
      A1[i][3] = (_Float16)b1[M];            // bias via constant-1 input row
    }
  }

  #pragma unroll
  for (int i = 0; i < 4; i++) {
    int M = 16 * i + m16;
    #pragma unroll
    for (int ks = 0; ks < 2; ks++) {
      v16h a = {};
      #pragma unroll
      for (int h = 0; h < 16; h++) {
        int K = 32 * ks + 8 * chunk + ((h < 8) ? h : (h + 8));
        a[h] = (_Float16)W2[K * HIDDEN + M];   // W2^T[M][K] = W2[K][M]
      }
      A2[i][ks] = a;
    }
  }

  #pragma unroll
  for (int ks = 0; ks < 2; ks++) {
    v16h a = {};
    if (m16 == 0) {
      #pragma unroll
      for (int h = 0; h < 16; h++) {
        int K = 32 * ks + 8 * chunk + ((h < 8) ? h : (h + 8));
        a[h] = (_Float16)W3[K];
      }
    }
    A3[ks] = a;
  }

  #pragma unroll
  for (int i = 0; i < 4; i++) {
    v8h c = {};
    #pragma unroll
    for (int h = 0; h < 8; h++) c[h] = (_Float16)b2[16 * i + 8 * chunk + h];
    C2b[i] = c;
  }

  // B1 fragment: only halves 0..3 on chunk-0 lanes ever become nonzero;
  // keep the structural zeros resident across all tiles/rays.
  v16h B1f = {};

  // ---------------- Grid-stride over rays: one ray per wave per iteration.
  #pragma unroll 1
  for (int ray = waveG; ray < N_RAYS; ray += TOTAL_WAVES) {
    if (ray + TOTAL_WAVES < N_RAYS) {      // hint next ray into cache
      __builtin_prefetch(&rays_o[(ray + TOTAL_WAVES) * 3], 0, 3);
      __builtin_prefetch(&rays_d[(ray + TOTAL_WAVES) * 3], 0, 3);
    }
    float ox = rays_o[ray * 3 + 0], oy = rays_o[ray * 3 + 1], oz = rays_o[ray * 3 + 2];
    float dx = rays_d[ray * 3 + 0], dy = rays_d[ray * 3 + 1], dz = rays_d[ray * 3 + 2];

    // ray/AABB slab test
    const float eps = 1e-8f;
    float sx = (fabsf(dx) < eps) ? ((dx >= 0.0f) ? eps : -eps) : dx;
    float sy = (fabsf(dy) < eps) ? ((dy >= 0.0f) ? eps : -eps) : dy;
    float sz = (fabsf(dz) < eps) ? ((dz >= 0.0f) ? eps : -eps) : dz;
    float ix = 1.0f / sx, iy = 1.0f / sy, iz = 1.0f / sz;
    float t0x = (-BOUNDV - ox) * ix, t1x = (BOUNDV - ox) * ix;
    float t0y = (-BOUNDV - oy) * iy, t1y = (BOUNDV - oy) * iy;
    float t0z = (-BOUNDV - oz) * iz, t1z = (BOUNDV - oz) * iz;
    float tnear = fmaxf(fmaxf(fminf(t0x, t1x), fminf(t0y, t1y)), fminf(t0z, t1z));
    float tfar  = fminf(fminf(fmaxf(t0x, t1x), fmaxf(t0y, t1y)), fmaxf(t0z, t1z));
    float nearT = fmaxf(tnear, MIN_NEAR);
    float farT  = fmaxf(tfar, nearT);

    float carry = 0.0f, opa = 0.0f, dep = 0.0f;

    #pragma unroll 1
    for (int t = 0; t < ntiles; t++) {
      // lane m16 owns sample s = 16t + m16 of this ray (N dim of the GEMMs)
      float tm = nearT + ((float)(t * 16 + m16) + 0.5f) * step;

      // B1 = posaug^T fragment: only lanes<16, K=0..3 nonzero -> (x,y,z,1)
      if (chunk == 0) {
        float px = fminf(fmaxf(ox + dx * tm, -BOUNDV), BOUNDV);
        float py = fminf(fmaxf(oy + dy * tm, -BOUNDV), BOUNDV);
        float pz = fminf(fmaxf(oz + dz * tm, -BOUNDV), BOUNDV);
        B1f[0] = (_Float16)px; B1f[1] = (_Float16)py;
        B1f[2] = (_Float16)pz; B1f[3] = (_Float16)1.0f;
      }

      // Layer 1: h1^T[64x16] = W1aug^T x posaug^T  (4 WMMAs, f16 accum)
      v8h c1[4];
      #pragma unroll
      for (int i = 0; i < 4; i++) { v8h z = {}; c1[i] = wmma_h(A1[i], B1f, z); }

      // handoff: B2[ks] = relu(concat(c1[2ks], c1[2ks+1]))
      v16h B2f0 = relu_cat(c1[0], c1[1]);
      v16h B2f1 = relu_cat(c1[2], c1[3]);

      // Layer 2: h2^T = W2^T x h1^T + b2 (8 WMMAs, bias as initial C)
      v8h c2[4];
      #pragma unroll
      for (int i = 0; i < 4; i++) {
        v8h acc = C2b[i];
        acc = wmma_h(A2[i][0], B2f0, acc);
        acc = wmma_h(A2[i][1], B2f1, acc);
        c2[i] = acc;
      }

      // handoff: B3[ks] = relu(concat(c2[2ks], c2[2ks+1]))
      v16h B3f0 = relu_cat(c2[0], c2[1]);
      v16h B3f1 = relu_cat(c2[2], c2[3]);

      // Layer 3: raw^T = W3^T x h2^T (2 WMMAs, f32 accum); lane=sample in c3[0]
      v8f c3 = {};
      c3 = wmma_f(A3[0], B3f0, c3);
      c3 = wmma_f(A3[1], B3f1, c3);

      float raw   = c3[0] + b3s;                      // valid on lanes 0..15
      float sigma = __expf(fminf(fmaxf(raw, -15.0f), 15.0f));
      bool  vs    = (tm < farT) && (lane < 16);
      float delta = vs ? sigma * step : 0.0f;

      // inclusive scan over the 16 sample lanes
      float incl = delta;
      #pragma unroll
      for (int off = 1; off < 16; off <<= 1) {
        float nv = __shfl_up(incl, off, 32);
        if (lane >= off) incl += nv;
      }
      float excl  = incl - delta;
      float trans = __expf(-(carry + excl));          // exclusive transmittance
      float alpha = 1.0f - __expf(-delta);            // alpha==0 on dead lanes -> w==0
      float w     = alpha * trans;
      opa += w;
      dep += w * tm;
      carry += __shfl(incl, 15, 32);                  // tile's total optical depth
    }

    // reduce opacity/depth over the 16 sample lanes (lanes>=16 carry zeros)
    #pragma unroll
    for (int off = 8; off >= 1; off >>= 1) {
      opa += __shfl_xor(opa, off, 32);
      dep += __shfl_xor(dep, off, 32);
    }
    if (lane == 0) {
      out[ray * 2 + 0] = opa;
      out[ray * 2 + 1] = dep;
    }
  }
}

extern "C" void kernel_launch(void* const* d_in, const int* in_sizes, int n_in,
                              void* d_out, int out_size, void* d_ws, size_t ws_size,
                              hipStream_t stream) {
  const float* rays_o = (const float*)d_in[0];
  const float* rays_d = (const float*)d_in[1];
  const float* W1     = (const float*)d_in[2];
  const float* b1     = (const float*)d_in[3];
  const float* W2     = (const float*)d_in[4];
  const float* b2     = (const float*)d_in[5];
  const float* W3     = (const float*)d_in[6];
  const float* b3     = (const float*)d_in[7];
  const int*   ns     = (const int*)d_in[8];
  float*       outp   = (float*)d_out;

  nerf_render_kernel<<<NUM_BLOCKS, 256, 0, stream>>>(
      rays_o, rays_d, W1, b1, W2, b2, W3, b3, ns, outp);
}
